// FlashAttention_41850161332976
// MI455X (gfx1250) — matrix operand order repaired and checked by reading, not compiled, not run on hardware
//
#include <hip/hip_runtime.h>
#include <hip/hip_bf16.h>

#define BATCH  2
#define SEQ    2048
#define DMODEL 1024
#define HEADS  16
#define DHEAD  64
#define INNER  1024

typedef __attribute__((ext_vector_type(8)))  float          v8f;
typedef __attribute__((ext_vector_type(16))) __bf16         v16bf;
typedef __attribute__((ext_vector_type(8)))  __bf16         v8bf;
typedef __attribute__((ext_vector_type(8)))  unsigned short v8u16;

__device__ __forceinline__ unsigned short f2bf(float f) {
  union { float f; unsigned int u; } c; c.f = f;
  unsigned int r = c.u + 0x7FFFu + ((c.u >> 16) & 1u);
  return (unsigned short)(r >> 16);
}

__device__ __forceinline__ v16bf cat8(v8bf lo, v8bf hi) {
  return __builtin_shufflevector(lo, hi, 0,1,2,3,4,5,6,7,8,9,10,11,12,13,14,15);
}

// A fragment (16x32 bf16): lane holds row m, K split-8 pattern per ISA 7.12.2
__device__ __forceinline__ v16bf load_a_frag(const __bf16* __restrict__ a, int ld,
                                             int m, int k0, int lane) {
  int koff = (lane & 16) ? 8 : 0;
  const __bf16* p = a + (size_t)m * ld + k0 + koff;
  v8bf lo = *(const v8bf*)p;          // K = k0+koff .. +7
  v8bf hi = *(const v8bf*)(p + 16);   // K = k0+16+koff .. +7
  return cat8(lo, hi);
}

// B fragment (32x16) from row-major Bt (rows = output columns):
// lane holds column n; K split-8 pattern (same as A fragment)
__device__ __forceinline__ v16bf load_bt_frag(const __bf16* __restrict__ bt, int ld,
                                              int n, int k0, int lane) {
  int koff = (lane & 16) ? 8 : 0;
  const __bf16* p = bt + (size_t)n * ld + k0 + koff;
  v8bf lo = *(const v8bf*)p;
  v8bf hi = *(const v8bf*)(p + 16);
  return cat8(lo, hi);
}

__device__ __forceinline__ v8f wmma_bf16(v16bf a, v16bf b, v8f c) {
  return __builtin_amdgcn_wmma_f32_16x16x32_bf16(false, a, false, b,
                                                 (short)0, c, false, false);
}

// ---------------- precision-convert kernels ----------------

__global__ void k_cast_bf16(const float* __restrict__ src,
                            unsigned short* __restrict__ dst, int n) {
  int i = blockIdx.x * blockDim.x + threadIdx.x;
  if (i < n) dst[i] = f2bf(src[i]);
}

// dst[c][r] = src[r][c]  (src: rows x cols row-major)
__global__ void k_cast_transpose_bf16(const float* __restrict__ src,
                                      unsigned short* __restrict__ dst,
                                      int rows, int cols) {
  int i = blockIdx.x * blockDim.x + threadIdx.x;
  if (i < rows * cols) {
    int r = i / cols, c = i - r * cols;
    dst[(size_t)c * rows + r] = f2bf(src[i]);
  }
}

// ---------------- QKV projection GEMM ----------------
// C[4096 x 3072] = xb[4096 x 1024] * Wqkvt^T ; one wave = 16x64 tile.
// cols 0..1023 -> Q[B,H,N,64]; 1024..2047 -> K; 2048..3071 -> V stored transposed [B,H,64,N]
__global__ void k_gemm_qkv(const __bf16* __restrict__ xb,
                           const __bf16* __restrict__ wt,
                           unsigned short* __restrict__ Qb,
                           unsigned short* __restrict__ Kb,
                           unsigned short* __restrict__ Vtb) {
  const int lane = threadIdx.x & 31;
  const int gw = blockIdx.x * (blockDim.x >> 5) + (threadIdx.x >> 5);
  const int cg = gw % 48;      // 64-col group
  const int rt = gw / 48;      // 16-row tile
  const int m  = rt * 16 + (lane & 15);

  v8f acc[4] = {};
  for (int k0 = 0; k0 < DMODEL; k0 += 32) {
    v16bf a = load_a_frag(xb, DMODEL, m, k0, lane);
#pragma unroll
    for (int t = 0; t < 4; ++t) {
      int n = cg * 64 + t * 16 + (lane & 15);
      acc[t] = wmma_bf16(a, load_bt_frag(wt, DMODEL, n, k0, lane), acc[t]);
    }
  }

  const int sec  = cg >> 4;          // 0=Q 1=K 2=V
  const int head = cg & 15;
  const int roff = (lane & 16) ? 8 : 0;
  const int b    = (rt * 16) / SEQ;
  const int n0   = (rt * 16) % SEQ;

  if (sec < 2) {
    unsigned short* dst = (sec == 0) ? Qb : Kb;
#pragma unroll
    for (int t = 0; t < 4; ++t) {
      int d = t * 16 + (lane & 15);
      size_t base = ((size_t)(b * HEADS + head) * SEQ + n0 + roff) * DHEAD + d;
#pragma unroll
      for (int r = 0; r < 8; ++r)
        dst[base + (size_t)r * DHEAD] = f2bf(acc[t][r]);
    }
  } else {
    // V transposed: contiguous 8xbf16 store per lane per tile
#pragma unroll
    for (int t = 0; t < 4; ++t) {
      int d = t * 16 + (lane & 15);
      size_t base = ((size_t)((b * HEADS + head) * DHEAD + d)) * SEQ + n0 + roff;
      v8u16 pk;
#pragma unroll
      for (int r = 0; r < 8; ++r) pk[r] = f2bf(acc[t][r]);
      *(v8u16*)(Vtb + base) = pk;
    }
  }
}

// ---------------- flash attention ----------------
// one wave = one (b,h,16-row query tile); key blocks of 32, causal
__global__ void k_attn(const __bf16* __restrict__ Q,
                       const __bf16* __restrict__ K,
                       const __bf16* __restrict__ Vt,
                       unsigned short* __restrict__ attn) {
  __shared__ unsigned short plds[8][16 * 32];
  const int lane = threadIdx.x & 31;
  const int w    = threadIdx.x >> 5;
  const int gw   = blockIdx.x * 8 + w;
  const int qt   = gw & 127;
  const int h    = (gw >> 7) & 15;
  const int b    = gw >> 11;
  const int roff = (lane & 16) ? 8 : 0;

  const __bf16* Qh = Q  + (size_t)(b * HEADS + h) * SEQ * DHEAD;
  const __bf16* Kh = K  + (size_t)(b * HEADS + h) * SEQ * DHEAD;
  const __bf16* Vh = Vt + (size_t)(b * HEADS + h) * DHEAD * SEQ;

  const int mrow = qt * 16 + (lane & 15);
  v16bf aq0 = load_a_frag(Qh, DHEAD, mrow, 0, lane);
  v16bf aq1 = load_a_frag(Qh, DHEAD, mrow, 32, lane);

  v8f o[4] = {};
  float mrun[8], lrun[8];
#pragma unroll
  for (int r = 0; r < 8; ++r) { mrun[r] = -3.0e38f; lrun[r] = 0.0f; }

  const float scale = 0.125f;  // 64^-0.5
  const int nkb = (qt * 16 + 15) / 32 + 1;

  for (int kb = 0; kb < nkb; ++kb) {
    const int j0 = kb * 32;
    // S = Q * K^T  (two 16x16 tiles)
    v8f s0 = {}, s1 = {};
    {
      int j = j0 + (lane & 15);
      s0 = wmma_bf16(aq0, load_bt_frag(Kh, DHEAD, j, 0, lane), s0);
      s0 = wmma_bf16(aq1, load_bt_frag(Kh, DHEAD, j, 32, lane), s0);
      int j2 = j0 + 16 + (lane & 15);
      s1 = wmma_bf16(aq0, load_bt_frag(Kh, DHEAD, j2, 0, lane), s1);
      s1 = wmma_bf16(aq1, load_bt_frag(Kh, DHEAD, j2, 32, lane), s1);
    }
#pragma unroll
    for (int r = 0; r < 8; ++r) { s0[r] *= scale; s1[r] *= scale; }

    if (j0 + 31 > qt * 16) {  // diagonal block: causal mask
      int jj0 = j0 + (lane & 15), jj1 = jj0 + 16;
#pragma unroll
      for (int r = 0; r < 8; ++r) {
        int i = qt * 16 + r + roff;
        if (jj0 > i) s0[r] = -3.0e38f;
        if (jj1 > i) s1[r] = -3.0e38f;
      }
    }

    // online softmax (row reductions within 16-lane halves)
    float c[8];
#pragma unroll
    for (int r = 0; r < 8; ++r) {
      float mx = fmaxf(s0[r], s1[r]);
      mx = fmaxf(mx, __shfl_xor(mx, 1));
      mx = fmaxf(mx, __shfl_xor(mx, 2));
      mx = fmaxf(mx, __shfl_xor(mx, 4));
      mx = fmaxf(mx, __shfl_xor(mx, 8));
      float mnew = fmaxf(mrun[r], mx);
      c[r] = __expf(mrun[r] - mnew);
      mrun[r] = mnew;
      float e0 = __expf(s0[r] - mnew);
      float e1 = __expf(s1[r] - mnew);
      s0[r] = e0; s1[r] = e1;
      float rs = e0 + e1;
      rs += __shfl_xor(rs, 1);
      rs += __shfl_xor(rs, 2);
      rs += __shfl_xor(rs, 4);
      rs += __shfl_xor(rs, 8);
      lrun[r] = lrun[r] * c[r] + rs;
    }
#pragma unroll
    for (int t = 0; t < 4; ++t)
#pragma unroll
      for (int r = 0; r < 8; ++r) o[t][r] *= c[r];

    // P: C-layout -> A-layout via per-wave LDS tile (16x32 bf16)
    unsigned short* pl = plds[w];
    {
      int col = lane & 15;
#pragma unroll
      for (int r = 0; r < 8; ++r) {
        int row = r + roff;
        pl[row * 32 + col]      = f2bf(s0[r]);
        pl[row * 32 + 16 + col] = f2bf(s1[r]);
      }
    }
    asm volatile("s_wait_dscnt 0x0" ::: "memory");
    v16bf ap;
    {
      int mm = lane & 15;
      v8bf plo = *(const v8bf*)(const void*)(pl + mm * 32 + roff);
      v8bf phi = *(const v8bf*)(const void*)(pl + mm * 32 + 16 + roff);
      ap = cat8(plo, phi);
    }

    // O += P * V  (V stored transposed: rows = d, contiguous over keys)
#pragma unroll
    for (int t = 0; t < 4; ++t) {
      int d = t * 16 + (lane & 15);
      o[t] = wmma_bf16(ap, load_bt_frag(Vh, SEQ, d, j0, lane), o[t]);
    }
  }

  // normalize + store attn out [B, N, H*64] as bf16
#pragma unroll
  for (int t = 0; t < 4; ++t) {
    int d = t * 16 + (lane & 15);
#pragma unroll
    for (int r = 0; r < 8; ++r) {
      int row = qt * 16 + r + roff;
      attn[((size_t)b * SEQ + row) * INNER + h * DHEAD + d] = f2bf(o[t][r] / lrun[r]);
    }
  }
}

// ---------------- output projection ----------------
__global__ void k_gemm_out(const __bf16* __restrict__ A,
                           const __bf16* __restrict__ Wot,
                           float* __restrict__ out) {
  const int lane = threadIdx.x & 31;
  const int gw = blockIdx.x * (blockDim.x >> 5) + (threadIdx.x >> 5);
  const int cg = gw % 16;
  const int rt = gw / 16;
  const int m  = rt * 16 + (lane & 15);

  v8f acc[4] = {};
  for (int k0 = 0; k0 < INNER; k0 += 32) {
    v16bf a = load_a_frag(A, INNER, m, k0, lane);
#pragma unroll
    for (int t = 0; t < 4; ++t) {
      int n = cg * 64 + t * 16 + (lane & 15);
      acc[t] = wmma_bf16(a, load_bt_frag(Wot, INNER, n, k0, lane), acc[t]);
    }
  }
  const int roff = (lane & 16) ? 8 : 0;
#pragma unroll
  for (int t = 0; t < 4; ++t) {
    int col = cg * 64 + t * 16 + (lane & 15);
#pragma unroll
    for (int r = 0; r < 8; ++r) {
      int row = rt * 16 + r + roff;
      out[(size_t)row * DMODEL + col] = acc[t][r];
    }
  }
}

extern "C" void kernel_launch(void* const* d_in, const int* in_sizes, int n_in,
                              void* d_out, int out_size, void* d_ws, size_t ws_size,
                              hipStream_t stream) {
  (void)in_sizes; (void)n_in; (void)out_size; (void)ws_size;
  const float* x   = (const float*)d_in[0];
  const float* Wq  = (const float*)d_in[1];
  const float* Wkv = (const float*)d_in[2];
  const float* Wo  = (const float*)d_in[3];
  float* out = (float*)d_out;

  unsigned short* ws = (unsigned short*)d_ws;
  size_t off = 0;
  unsigned short* xb    = ws + off; off += (size_t)BATCH * SEQ * DMODEL;       // 4M
  unsigned short* wqkvt = ws + off; off += (size_t)3 * INNER * DMODEL;         // 3M
  unsigned short* wot   = ws + off; off += (size_t)DMODEL * INNER;             // 1M
  unsigned short* Qb    = ws + off; off += (size_t)BATCH * HEADS * SEQ * DHEAD;
  unsigned short* Kb    = ws + off; off += (size_t)BATCH * HEADS * SEQ * DHEAD;
  unsigned short* Vtb   = ws + off; off += (size_t)BATCH * HEADS * DHEAD * SEQ;
  unsigned short* attn  = ws + off; off += (size_t)BATCH * SEQ * INNER;

  const int nx = BATCH * SEQ * DMODEL;
  k_cast_bf16<<<(nx + 255) / 256, 256, 0, stream>>>(x, xb, nx);
  k_cast_transpose_bf16<<<(DMODEL * INNER + 255) / 256, 256, 0, stream>>>(
      Wq, wqkvt, DMODEL, INNER);
  k_cast_transpose_bf16<<<(DMODEL * 2 * INNER + 255) / 256, 256, 0, stream>>>(
      Wkv, wqkvt + (size_t)INNER * DMODEL, DMODEL, 2 * INNER);
  k_cast_transpose_bf16<<<(INNER * DMODEL + 255) / 256, 256, 0, stream>>>(
      Wo, wot, INNER, DMODEL);

  // 256 row-tiles * 48 col-groups = 12288 waves / 8 per block
  k_gemm_qkv<<<1536, 256, 0, stream>>>((const __bf16*)xb, (const __bf16*)wqkvt,
                                       Qb, Kb, Vtb);
  // 2*16*128 = 4096 waves / 8 per block
  k_attn<<<512, 256, 0, stream>>>((const __bf16*)Qb, (const __bf16*)Kb,
                                  (const __bf16*)Vtb, attn);
  // 256 row-tiles * 16 col-groups = 4096 waves / 8 per block
  k_gemm_out<<<512, 256, 0, stream>>>((const __bf16*)attn, (const __bf16*)wot, out);
}
